// MultiHeadAttention_9792525434968
// MI455X (gfx1250) — matrix operand order repaired and checked by reading, not compiled, not run on hardware
//
#include <hip/hip_runtime.h>
#include <hip/hip_bf16.h>

// ---------------------------------------------------------------------------
// Fused causal MHA for MI455X (gfx1250, wave32, WMMA + TDM + async-LDS).
// Pipeline: fp32->bf16 convert | QKV GEMM (wmma bf16, TDM A-tile) |
//           flash attention (wmma, TDM K-tile, async Q-tile) |
//           out-proj GEMM (wmma bf16, fp32 out).
// ---------------------------------------------------------------------------

typedef __bf16 bf16_t;
typedef __attribute__((ext_vector_type(8)))  bf16_t bf16x8;
typedef __attribute__((ext_vector_type(16))) bf16_t bf16x16;
typedef __attribute__((ext_vector_type(8)))  float  f32x8;
typedef __attribute__((ext_vector_type(4)))  unsigned int u32x4;
typedef __attribute__((ext_vector_type(4)))  int i32x4;
typedef __attribute__((ext_vector_type(8)))  int i32x8;

constexpr int Bsz  = 2;
constexpr int Tseq = 2048;
constexpr int Dm   = 1024;
constexpr int Hn   = 16;
constexpr int Dh   = 64;               // head dim
constexpr int Mrows = Bsz * Tseq;      // 4096 token rows

#if __has_builtin(__builtin_amdgcn_tensor_load_to_lds)
#define ATH_HAVE_TDM 1
#else
#define ATH_HAVE_TDM 0
#endif

__device__ __forceinline__ void ath_wait_tensorcnt0() {
#if __has_builtin(__builtin_amdgcn_s_wait_tensorcnt)
    __builtin_amdgcn_s_wait_tensorcnt(0);
#else
    asm volatile("s_wait_tensorcnt 0x0" ::: "memory");
#endif
}

__device__ __forceinline__ void ath_wait_asynccnt0() {
#if __has_builtin(__builtin_amdgcn_s_wait_asynccnt)
    __builtin_amdgcn_s_wait_asynccnt(0);
#else
    asm volatile("s_wait_asynccnt 0x0" ::: "memory");
#endif
}

// Per-lane async global->LDS 16B copy (VGLOBAL encoding, GV mode, ASYNCcnt).
__device__ __forceinline__ void ath_async_b128(unsigned lds_byte_off, const void* g) {
    asm volatile("global_load_async_to_lds_b128 %0, %1, off"
                 :: "v"(lds_byte_off), "v"((unsigned long long)(uintptr_t)g)
                 : "memory");
}

#if ATH_HAVE_TDM
// 2D TDM load: tile_h rows x tile_w bf16 elements, global row stride
// `stride_el` elements, into LDS at byte offset lds_off (row-major, packed).
// D# layout per CDNA5 ISA 8.3-8.4: group0 {count=1, lds_addr, global_addr,
// type=2}; group1 {data_size=1(2B), tensor dims, tile dims, dim0 stride}.
__device__ __forceinline__ void ath_tdm_load_2d(unsigned lds_off, const void* g,
                                                unsigned tile_w, unsigned tile_h,
                                                unsigned long long stride_el) {
    unsigned long long ga = (unsigned long long)(uintptr_t)g;
    u32x4 g0;
    g0[0] = 1u;                                              // count=1, is_restore=0
    g0[1] = lds_off;                                         // lds_addr (bytes)
    g0[2] = (unsigned)ga;                                    // global_addr[31:0]
    g0[3] = (unsigned)((ga >> 32) & 0x01FFFFFFu) | (2u << 30); // [56:32] + type=2
    i32x8 g1;
    g1[0] = (int)(1u << 16);                                 // data_size=1 (2 bytes)
    g1[1] = (int)((tile_w & 0xFFFFu) << 16);                 // tensor_dim0[15:0]
    g1[2] = (int)((tile_w >> 16) | ((tile_h & 0xFFFFu) << 16)); // dim0 hi | dim1 lo
    g1[3] = (int)((tile_h >> 16) | (tile_w << 16));          // dim1 hi | tile_dim0
    g1[4] = (int)(tile_h & 0xFFFFu);                         // tile_dim1 (tile_dim2=0)
    g1[5] = (int)(unsigned)(stride_el & 0xFFFFFFFFu);        // dim0_stride[31:0]
    g1[6] = (int)(unsigned)((stride_el >> 32) & 0xFFFFu);    // dim0_stride[47:32]
    g1[7] = 0;                                               // dim1_stride unused (2D)
    i32x4 gz = {0, 0, 0, 0};
#if __clang_major__ >= 23
    i32x8 gz8 = {0, 0, 0, 0, 0, 0, 0, 0};
    __builtin_amdgcn_tensor_load_to_lds(g0, g1, gz, gz, gz8, 0);
#else
    __builtin_amdgcn_tensor_load_to_lds(g0, g1, gz, gz, 0);
#endif
}
#endif // ATH_HAVE_TDM

__device__ __forceinline__ f32x8 zero8() {
    f32x8 z;
#pragma unroll
    for (int i = 0; i < 8; ++i) z[i] = 0.0f;
    return z;
}

// D = A(16x32 bf16) * B(32x16 bf16) + C(16x16 f32) on one wave32.
__device__ __forceinline__ f32x8 wmma_bf16(bf16x16 a, bf16x16 b, f32x8 c) {
    return __builtin_amdgcn_wmma_f32_16x16x32_bf16(
        false, a, false, b, (short)0, c, false, false);
}

// Build a 16-element bf16 fragment from two contiguous 16B LDS chunks.
__device__ __forceinline__ bf16x16 ld_frag(const bf16_t* p0, const bf16_t* p1) {
    bf16x8 lo = *reinterpret_cast<const bf16x8*>(p0);
    bf16x8 hi = *reinterpret_cast<const bf16x8*>(p1);
    bf16x16 r;
#pragma unroll
    for (int i = 0; i < 8; ++i) { r[i] = lo[i]; r[i + 8] = hi[i]; }
    return r;
}

// ---------------------------------------------------------------------------
// fp32 -> bf16 converter (vectorized x4)
// ---------------------------------------------------------------------------
__global__ void __launch_bounds__(256)
cvt_f32_bf16(const float* __restrict__ in, bf16_t* __restrict__ out, int n4) {
    int i = blockIdx.x * blockDim.x + threadIdx.x;
    if (i < n4) {
        float4 f = reinterpret_cast<const float4*>(in)[i];
        union { bf16_t h[4]; uint2 u; } pk;
        pk.h[0] = (bf16_t)f.x; pk.h[1] = (bf16_t)f.y;
        pk.h[2] = (bf16_t)f.z; pk.h[3] = (bf16_t)f.w;
        reinterpret_cast<uint2*>(out)[i] = pk.u;
    }
}

// ---------------------------------------------------------------------------
// GEMM: C[M,N] = A[M,K](bf16) * B[K,N](bf16) + bias[N]
// Block tile 128x128, K-step 32. A-tile filled by one TDM DMA per K-step
// (TENSORcnt); B-tile staged transposed. 8 waves of 32x64 (2x4 WMMA frags).
// ---------------------------------------------------------------------------
template <typename OutT>
__global__ void __launch_bounds__(256)
gemm_bf16_bias(const bf16_t* __restrict__ A, const bf16_t* __restrict__ Bm,
               const float* __restrict__ bias, OutT* __restrict__ C,
               int M, int N, int K) {
    constexpr int BM = 128, BN = 128, BK = 32;
    __shared__ __align__(16) struct {
        bf16_t As[BM * BK];      // As[m*BK + k]   (8 KB, LDS byte offset 0)
        bf16_t Bs[BN * BK];      // Bs[n*BK + k]   (8 KB, LDS byte offset 8192)
    } sm;

    const int tid  = threadIdx.x;
    const int lane = tid & 31;
    const int wave = tid >> 5;
    const int ln   = lane & 15;
    const int hih  = lane >> 4;
    const int wm   = wave >> 1;
    const int wn   = wave & 1;
    const int row0 = blockIdx.y * BM;
    const int col0 = blockIdx.x * BN;

    const int kb = hih ? 8 : 0;              // unused; A fragment also takes K base kh
    const int kh = hih ? 16 : 0;             // B-fragment K base

    f32x8 acc[2][4];
#pragma unroll
    for (int mi = 0; mi < 2; ++mi)
#pragma unroll
        for (int ni = 0; ni < 4; ++ni) acc[mi][ni] = zero8();

    for (int k0 = 0; k0 < K; k0 += BK) {
        // ---- A tile: one TDM DMA (wave 0) or per-lane copy fallback ----
#if ATH_HAVE_TDM
        if (wave == 0)
            ath_tdm_load_2d(/*lds_off=*/0u, &A[(size_t)row0 * K + k0],
                            /*tile_w=*/BK, /*tile_h=*/BM, /*stride=*/(unsigned long long)K);
#else
#pragma unroll
        for (int i = 0; i < 2; ++i) {
            int chunk = tid + i * 256;
            int r = chunk >> 2;
            int c = (chunk & 3) * 8;
            *reinterpret_cast<uint4*>(&sm.As[r * BK + c]) =
                *reinterpret_cast<const uint4*>(&A[(size_t)(row0 + r) * K + k0 + c]);
        }
#endif
        // ---- B tile: global 16B loads, transposed scatter into LDS ----
#pragma unroll
        for (int i = 0; i < 2; ++i) {
            int chunk = tid + i * 256;                     // 0..511
            int kk = chunk >> 4;                           // 16 chunks per k-row
            int nn = (chunk & 15) * 8;
            uint4 raw = *reinterpret_cast<const uint4*>(
                &Bm[(size_t)(k0 + kk) * N + col0 + nn]);
            bf16x8 v = __builtin_bit_cast(bf16x8, raw);
#pragma unroll
            for (int e = 0; e < 8; ++e) sm.Bs[(nn + e) * BK + kk] = v[e];
        }
        if (k0 + BK < K)      // gfx1250 global_prefetch_b8 of next B tile
            __builtin_prefetch(&Bm[(size_t)(k0 + BK + (tid >> 4)) * N + col0], 0, 1);
#if ATH_HAVE_TDM
        if (wave == 0) ath_wait_tensorcnt0();
#endif
        __syncthreads();

        // ---- fragments + 8 WMMAs ----
        bf16x16 af[2];
#pragma unroll
        for (int mi = 0; mi < 2; ++mi) {
            const bf16_t* p = &sm.As[(wm * 32 + mi * 16 + ln) * BK];
            af[mi] = ld_frag(p + kh, p + kh + 8);
        }
#pragma unroll
        for (int ni = 0; ni < 4; ++ni) {
            const bf16_t* p = &sm.Bs[(wn * 64 + ni * 16 + ln) * BK + kh];
            bf16x16 bfv = ld_frag(p, p + 8);
#pragma unroll
            for (int mi = 0; mi < 2; ++mi)
                acc[mi][ni] = wmma_bf16(af[mi], bfv, acc[mi][ni]);
        }
        __syncthreads();
    }

    // ---- epilogue: bias add, store (C-fragment layout: row = j + 8*hih) ----
#pragma unroll
    for (int mi = 0; mi < 2; ++mi)
#pragma unroll
        for (int ni = 0; ni < 4; ++ni) {
            int c = col0 + wn * 64 + ni * 16 + ln;
            float bv = bias[c];
#pragma unroll
            for (int j = 0; j < 8; ++j) {
                int r = row0 + wm * 32 + mi * 16 + j + 8 * hih;
                C[(size_t)r * N + c] = static_cast<OutT>(acc[mi][ni][j] + bv);
            }
        }
}

// ---------------------------------------------------------------------------
// Flash attention. qkv: bf16 [B*T, 3D]. Grid: (T/128, B*H), 256 threads.
// Q tile via async global->LDS (ASYNCcnt); K tile via TDM (TENSORcnt);
// V tile transposed manually. Each wave owns a 16-row query strip.
// ---------------------------------------------------------------------------
__global__ void __launch_bounds__(256)
flash_attn(const bf16_t* __restrict__ qkv, bf16_t* __restrict__ ctx) {
    const int tid  = threadIdx.x;
    const int lane = tid & 31;
    const int wave = tid >> 5;
    const int ln   = lane & 15;
    const int hih  = lane >> 4;
    const int kb   = hih ? 8 : 0;
    const int kh   = hih ? 16 : 0;

    const int qt = blockIdx.x * 128;
    const int bb = blockIdx.y / Hn;
    const int hh = blockIdx.y % Hn;

    __shared__ __align__(16) struct {
        bf16_t Qs[128 * Dh];     // 16 KB, LDS byte off 0
        bf16_t Ks[64 * Dh];      //  8 KB, LDS byte off 16384
        bf16_t Vt[Dh * 64];      //  8 KB, LDS byte off 24576 (Vt[d*64+j])
        bf16_t Ps[8 * 16 * 64];  // 16 KB, LDS byte off 32768
    } sm;
    constexpr unsigned KS_OFF = 16384u;

    const size_t rs = (size_t)3 * Dm;       // qkv row stride (elements)
    const bf16_t* qb  = qkv + (size_t)bb * Tseq * rs + 0 * Dm + hh * Dh;
    const bf16_t* kbp = qkv + (size_t)bb * Tseq * rs + 1 * Dm + hh * Dh;
    const bf16_t* vbp = qkv + (size_t)bb * Tseq * rs + 2 * Dm + hh * Dh;

    // ---- Q tile (128 x 64): async global->LDS b128 per lane ----
#pragma unroll
    for (int i = 0; i < 4; ++i) {
        int chunk = tid + i * 256;          // 0..1023, 8 chunks per row
        int r = chunk >> 3;
        int c = (chunk & 7) * 8;
        ath_async_b128((unsigned)((r * Dh + c) * 2),
                       &qb[(size_t)(qt + r) * rs + c]);
    }
    ath_wait_asynccnt0();
    __syncthreads();

    bf16x16 aq[2];
    {
        const bf16_t* p = &sm.Qs[(wave * 16 + ln) * Dh];
        aq[0] = ld_frag(p + kh,      p + kh + 8);
        aq[1] = ld_frag(p + 32 + kh, p + 32 + kh + 8);
    }

    f32x8 acc[4];
#pragma unroll
    for (int ni = 0; ni < 4; ++ni) acc[ni] = zero8();
    float m_i[8], l_i[8];
#pragma unroll
    for (int j = 0; j < 8; ++j) { m_i[j] = -1e30f; l_i[j] = 0.0f; }

    const float sm_scale = 0.125f;          // 1/sqrt(64)
    const int nkt = (qt >> 6) + 2;          // causal tile bound

    for (int kt = 0; kt < nkt; ++kt) {
        __syncthreads();                    // previous-iter readers done
        // ---- K tile (64x64): one TDM DMA (wave 0) or copy fallback ----
#if ATH_HAVE_TDM
        if (wave == 0)
            ath_tdm_load_2d(KS_OFF, &kbp[(size_t)(kt * 64) * rs],
                            /*tile_w=*/Dh, /*tile_h=*/64,
                            /*stride=*/(unsigned long long)rs);
#else
#pragma unroll
        for (int i = 0; i < 2; ++i) {
            int chunk = tid + i * 256;
            int r = chunk >> 3;
            int c = (chunk & 7) * 8;
            *reinterpret_cast<uint4*>(&sm.Ks[r * Dh + c]) =
                *reinterpret_cast<const uint4*>(&kbp[(size_t)(kt * 64 + r) * rs + c]);
        }
#endif
        // ---- V tile transposed: Vt[d][j] = V[j][d] ----
#pragma unroll
        for (int i = 0; i < 2; ++i) {
            int chunk = tid + i * 256;
            int r = chunk >> 3;             // key row j
            int c = (chunk & 7) * 8;        // first d
            uint4 raw = *reinterpret_cast<const uint4*>(
                &vbp[(size_t)(kt * 64 + r) * rs + c]);
            bf16x8 v = __builtin_bit_cast(bf16x8, raw);
#pragma unroll
            for (int e = 0; e < 8; ++e) sm.Vt[(c + e) * 64 + r] = v[e];
        }
#if ATH_HAVE_TDM
        if (wave == 0) ath_wait_tensorcnt0();
#endif
        __syncthreads();

        // ---- S strip (16x64) = Q_strip * K^T : 8 WMMAs ----
        f32x8 s[4];
#pragma unroll
        for (int ni = 0; ni < 4; ++ni) {
            const bf16_t* p = &sm.Ks[(ni * 16 + ln) * Dh];
            bf16x16 b0 = ld_frag(p + kh,      p + kh + 8);
            bf16x16 b1 = ld_frag(p + 32 + kh, p + 32 + kh + 8);
            s[ni] = zero8();
            s[ni] = wmma_bf16(aq[0], b0, s[ni]);
            s[ni] = wmma_bf16(aq[1], b1, s[ni]);
        }

        // ---- scale, causal mask, online softmax (wave32 C-layout aware) ----
        float alpha[8];
#pragma unroll
        for (int j = 0; j < 8; ++j) {
            int qrow = qt + wave * 16 + j + 8 * hih;
            float mx = -1e30f;
#pragma unroll
            for (int ni = 0; ni < 4; ++ni) {
                int kcol = kt * 64 + ni * 16 + ln;
                float v = s[ni][j] * sm_scale;
                v = (kcol <= qrow) ? v : -1e30f;
                s[ni][j] = v;
                mx = fmaxf(mx, v);
            }
#pragma unroll
            for (int off = 1; off < 16; off <<= 1)     // halves reduce separately
                mx = fmaxf(mx, __shfl_xor(mx, off, 32));
            float mn = fmaxf(m_i[j], mx);
            alpha[j] = __expf(m_i[j] - mn);
            m_i[j] = mn;
            float rsum = 0.0f;
#pragma unroll
            for (int ni = 0; ni < 4; ++ni) {
                float p = __expf(s[ni][j] - mn);
                s[ni][j] = p;
                rsum += p;
            }
#pragma unroll
            for (int off = 1; off < 16; off <<= 1)
                rsum += __shfl_xor(rsum, off, 32);
            l_i[j] = l_i[j] * alpha[j] + rsum;
#pragma unroll
            for (int ni = 0; ni < 4; ++ni) acc[ni][j] *= alpha[j];
        }

        // ---- restage P (C-layout f32 -> A-layout bf16) via wave-private LDS
        bf16_t* ps = &sm.Ps[wave * 16 * 64];
#pragma unroll
        for (int ni = 0; ni < 4; ++ni)
#pragma unroll
            for (int j = 0; j < 8; ++j)
                ps[(j + 8 * hih) * 64 + ni * 16 + ln] = (bf16_t)s[ni][j];
        // per-wave LDS ops are in order (DScnt) -> no barrier needed

        bf16x16 ap[2];
        {
            const bf16_t* p = &ps[ln * 64];
            ap[0] = ld_frag(p + kh,      p + kh + 8);
            ap[1] = ld_frag(p + 32 + kh, p + 32 + kh + 8);
        }

        // ---- O += P * V : 8 WMMAs ----
#pragma unroll
        for (int ni = 0; ni < 4; ++ni) {
            const bf16_t* p = &sm.Vt[(ni * 16 + ln) * 64];
            bf16x16 b0 = ld_frag(p + kh,      p + kh + 8);
            bf16x16 b1 = ld_frag(p + 32 + kh, p + 32 + kh + 8);
            acc[ni] = wmma_bf16(ap[0], b0, acc[ni]);
            acc[ni] = wmma_bf16(ap[1], b1, acc[ni]);
        }
    }

    // ---- normalize and store context strip as bf16 [B*T, D] ----
#pragma unroll
    for (int j = 0; j < 8; ++j) l_i[j] = 1.0f / l_i[j];
    bf16_t* cb = ctx + (size_t)(bb * Tseq + qt + wave * 16) * Dm + hh * Dh;
#pragma unroll
    for (int ni = 0; ni < 4; ++ni)
#pragma unroll
        for (int j = 0; j < 8; ++j) {
            int r = j + 8 * hih;
            int c = ni * 16 + ln;
            cb[(size_t)r * Dm + c] = (bf16_t)(acc[ni][j] * l_i[j]);
        }
}

// ---------------------------------------------------------------------------
// Launch
// ---------------------------------------------------------------------------
extern "C" void kernel_launch(void* const* d_in, const int* in_sizes, int n_in,
                              void* d_out, int out_size, void* d_ws, size_t ws_size,
                              hipStream_t stream) {
    (void)in_sizes; (void)n_in; (void)out_size; (void)ws_size;

    const float* x     = (const float*)d_in[0];   // [B,T,D]
    const float* w_qkv = (const float*)d_in[1];   // [D,3D]
    const float* b_qkv = (const float*)d_in[2];   // [3D]
    const float* w_out = (const float*)d_in[3];   // [D,D]
    const float* b_out = (const float*)d_in[4];   // [D]
    float* out = (float*)d_out;                   // [B,T,D]

    char* ws = (char*)d_ws;
    size_t off = 0;
    auto carve = [&](size_t bytes) -> char* {
        char* p = ws + off;
        off = (off + bytes + 255) & ~(size_t)255;
        return p;
    };
    bf16_t* xb   = (bf16_t*)carve((size_t)Mrows * Dm * 2);        //  8 MB
    bf16_t* wqb  = (bf16_t*)carve((size_t)Dm * 3 * Dm * 2);       //  6 MB
    bf16_t* wob  = (bf16_t*)carve((size_t)Dm * Dm * 2);           //  2 MB
    bf16_t* qkvb = (bf16_t*)carve((size_t)Mrows * 3 * Dm * 2);    // 24 MB
    bf16_t* ctxb = (bf16_t*)carve((size_t)Mrows * Dm * 2);        //  8 MB

    // 1) downconvert fp32 -> bf16
    {
        int n4 = Mrows * Dm / 4;
        cvt_f32_bf16<<<n4 / 256, 256, 0, stream>>>(x, xb, n4);
        n4 = Dm * 3 * Dm / 4;
        cvt_f32_bf16<<<n4 / 256, 256, 0, stream>>>(w_qkv, wqb, n4);
        n4 = Dm * Dm / 4;
        cvt_f32_bf16<<<n4 / 256, 256, 0, stream>>>(w_out, wob, n4);
    }
    // 2) fused QKV projection: [4096,1024] x [1024,3072] -> bf16 qkv
    gemm_bf16_bias<bf16_t><<<dim3(3 * Dm / 128, Mrows / 128), 256, 0, stream>>>(
        xb, wqb, b_qkv, qkvb, Mrows, 3 * Dm, Dm);
    // 3) flash attention -> bf16 context [4096,1024]
    flash_attn<<<dim3(Tseq / 128, Bsz * Hn), 256, 0, stream>>>(qkvb, ctxb);
    // 4) output projection: [4096,1024] x [1024,1024] + bias -> fp32 out
    gemm_bf16_bias<float><<<dim3(Dm / 128, Mrows / 128), 256, 0, stream>>>(
        ctxb, wob, b_out, out, Mrows, Dm, Dm);
}
